// RobertaHybridEncoder_81363860456157
// MI455X (gfx1250) — matrix operand order repaired and checked by reading, not compiled, not run on hardware
//
#include <hip/hip_runtime.h>
#include <hip/hip_bf16.h>
#include <stdint.h>
#include <stddef.h>

// ---------------------------------------------------------------------------
// CDNA5 / gfx1250 (MI455X), wave32.
// Matrix path : v_wmma_f32_16x16x32_bf16 (fp32 accumulate).
// Data moving : Tensor Data Mover (TENSOR_LOAD_TO_LDS + s_wait_tensorcnt)
//               stages packed-weight slabs into LDS, double buffered.
// ---------------------------------------------------------------------------

typedef __attribute__((ext_vector_type(16))) __bf16        v16bf;
typedef __attribute__((ext_vector_type(8)))  float         v8f;
typedef __attribute__((ext_vector_type(4)))  unsigned int  u32x4;

union BFrag {                 // one WMMA bf16 operand fragment (8 VGPRs/lane)
  v16bf          v;
  u32x4          q[2];
  unsigned short s[16];
};

__device__ __forceinline__ unsigned short f2bf(float x) {
  // round-to-nearest-even fp32 -> bf16
  unsigned int u = __float_as_uint(x);
  return (unsigned short)((u + 0x7FFFu + ((u >> 16) & 1u)) >> 16);
}

static constexpr int B_ = 2, S_ = 512, H_ = 1024, D_ = 24;
static constexpr int OUT_HID = 1024;   // value-MLP hidden
static constexpr int VAL_IN  = 2 * H_; // 2048

// ---------------------------------------------------------------------------
// Tensor Data Mover: 1-D tile copy global -> LDS (data_size = 8 bytes).
// D# built per CDNA5 ISA (group0/group1; rest = 0:
// a <=2-D tensor leaves D# groups 2/3 NULL).
// This toolchain's builtin arity is the 6-arg clang-23 form:
//   (u32x4 g0, i32x8 g1, i32x4, i32x4, i32x8, i32 cpol)
// ---------------------------------------------------------------------------
#if defined(__has_builtin)
#  if __has_builtin(__builtin_amdgcn_tensor_load_to_lds) && \
      __has_builtin(__builtin_amdgcn_s_wait_tensorcnt)
#    define USE_TDM 1
#  endif
#endif
#ifndef USE_TDM
#  define USE_TDM 0
#endif

#if USE_TDM
typedef __attribute__((ext_vector_type(4))) unsigned int tdm_g0_t;
typedef __attribute__((ext_vector_type(8))) int          tdm_g1_t;
typedef __attribute__((ext_vector_type(4))) int          tdm_v4_t;
typedef __attribute__((ext_vector_type(8))) int          tdm_v8_t;

__device__ __forceinline__ void tdm_load_1d(unsigned lds_off,
                                            const void* gsrc,
                                            unsigned n8 /* # of 8B elems */) {
  const unsigned long long ga = (unsigned long long)(size_t)gsrc;
  tdm_g0_t g0;
  g0[0] = 1u;                                   // count=1 : valid user D#
  g0[1] = lds_off;                              // lds_addr (bytes)
  g0[2] = (unsigned)ga;                         // global_addr[31:0]
  g0[3] = (unsigned)((ga >> 32) & 0x01FFFFFFu)  // global_addr[56:32]
        | (2u << 30);                           // type = 2 ("image")
  tdm_g1_t g1;
  g1[0] = (int)(3u << 16);                      // data_size=3 (8B); mask/flags=0
  g1[1] = (int)((n8 & 0xFFFFu) << 16);          // tensor_dim0[15:0]  (bits 63:48)
  g1[2] = (int)(((n8 >> 16) & 0xFFFFu)          // tensor_dim0[31:16] (bits 79:64)
        | (1u << 16));                          // tensor_dim1 = 1
  g1[3] = (int)((n8 & 0xFFFFu) << 16);          // tile_dim0 = n8    (bits 127:112)
  g1[4] = 0;                                    // tile_dim1 = tile_dim2 = 0
  g1[5] = (int)n8;                              // tensor_dim0_stride[31:0]
  g1[6] = 0;
  g1[7] = 0;
  const tdm_v4_t z4 = {0, 0, 0, 0};
  const tdm_v8_t z8 = {0, 0, 0, 0, 0, 0, 0, 0};
  __builtin_amdgcn_tensor_load_to_lds(g0, g1, z4, z4, z8, 0);
}
#endif

// ---------------------------------------------------------------------------
// Kernel 1: Zj = H_j @ p_j, Zi = H_i @ p_i  (+ bf16 copy of H_j for GEMM1 A)
// ---------------------------------------------------------------------------
__global__ void __launch_bounds__(256)
proj_kernel(const float* __restrict__ Hj, const float* __restrict__ Hi,
            const float* __restrict__ pj, const float* __restrict__ pi,
            float* __restrict__ Zj, float* __restrict__ Zi,
            unsigned short* __restrict__ Hjb) {
  __shared__ float rowj[H_];
  __shared__ float rowi[H_];
  const int row = blockIdx.x;            // b*S + s
  const int tid = threadIdx.x;
  const float* hj = Hj + (size_t)row * H_;
  const float* hi = Hi + (size_t)row * H_;
  for (int h = tid; h < H_; h += 256) {
    float a = hj[h], c = hi[h];
    rowj[h] = a;
    rowi[h] = c;
    Hjb[(size_t)row * H_ + h] = f2bf(a);
  }
  __syncthreads();
  if (tid < D_) {
    float acc = 0.f;
    for (int h = 0; h < H_; ++h) acc += rowj[h] * pj[h * D_ + tid];
    Zj[row * D_ + tid] = acc;
  } else if (tid >= 32 && tid < 32 + D_) {
    const int d = tid - 32;
    float acc = 0.f;
    for (int h = 0; h < H_; ++h) acc += rowi[h] * pi[h * D_ + d];
    Zi[row * D_ + d] = acc;
  }
}

// ---------------------------------------------------------------------------
// Kernel 2: repack fp32 weight [K,N] into bf16 WMMA B-fragment order:
//   for (ktile=k/32, ntile=n/16, lane) 16 contiguous bf16 =
//   W[ktile*32 + 8*(lane/16) + (i&7) + 16*(i>>3)][ntile*16 + lane%16], i=0..15
// ---------------------------------------------------------------------------
__global__ void __launch_bounds__(256)
pack_kernel(const float* __restrict__ W, unsigned short* __restrict__ out,
            int K, int N) {
  const int tid   = blockIdx.x * 256 + threadIdx.x;
  const int total = (K >> 5) * (N >> 4) * 32;
  if (tid >= total) return;
  const int lane  = tid & 31;
  const int tile  = tid >> 5;
  const int ntn   = N >> 4;
  const int ntile = tile % ntn;
  const int ktile = tile / ntn;
  const int kb    = ktile * 32 + ((lane >> 4) << 3);
  const int n     = ntile * 16 + (lane & 15);
  const size_t ob = (size_t)tid * 16;
  for (int i = 0; i < 16; ++i)
    out[ob + i] = f2bf(W[(size_t)(kb + (i & 7) + ((i >> 3) << 4)) * N + n]);
}

// ---------------------------------------------------------------------------
// Kernel 3: fused pair-MLP (WMMA) + softmax + context, one block per (b, j).
// ---------------------------------------------------------------------------
__global__ void __launch_bounds__(256)
pair_kernel(const float* __restrict__ Zjg, const float* __restrict__ Zig,
            const unsigned short* __restrict__ W1p,
            const float* __restrict__ b1, const float* __restrict__ W2,
            const float* __restrict__ b2, const float* __restrict__ mask,
            const float* __restrict__ Hi, unsigned short* __restrict__ ctxb) {
  __shared__ u32x4 sW1[1152];   // packed W1: 3 kt * 6 nt * 32 lanes * 32B = 18KB
  __shared__ float sZj[D_];
  __shared__ float slog[S_];
  __shared__ float sred[256];

  const int tid  = threadIdx.x;
  const int wave = tid >> 5, lane = tid & 31, lh = lane >> 4, ln = lane & 15;
  const int b = blockIdx.x / S_, j = blockIdx.x % S_;
  const int rowj = b * S_ + j;

  // stage packed W1 (TDM when available) + this query's Zj row
#if USE_TDM
  if (wave == 0) {
    tdm_load_1d((unsigned)(size_t)(void*)sW1, W1p, 1152u * 16u / 8u);
    __builtin_amdgcn_s_wait_tensorcnt(0);
  }
#else
  {
    const u32x4* W1q = (const u32x4*)W1p;
    for (int i = tid; i < 1152; i += 256) sW1[i] = W1q[i];
  }
#endif
  if (tid < D_) sZj[tid] = Zjg[rowj * D_ + tid];
  __syncthreads();

  // ---- pair MLP: 32 key tiles of 16, round-robin across the 8 waves ----
  for (int t = wave; t < 32; t += 8) {
    const int i0   = t * 16;
    const int ikey = i0 + ln;                               // M row = key index
    const float* zi = Zig + (size_t)(b * S_ + ikey) * D_;

    v8f acc[6];
    const v8f vz = {0.f, 0.f, 0.f, 0.f, 0.f, 0.f, 0.f, 0.f};
#pragma unroll
    for (int nt = 0; nt < 6; ++nt) acc[nt] = vz;

#pragma unroll
    for (int kt = 0; kt < 3; ++kt) {
      // A fragment per 16-bit 16x32 layout: element e holds
      // K = kt*32 + (e&7) + 16*(e>>3) + 8*(lane/16)
      BFrag a;
#pragma unroll
      for (int e = 0; e < 16; ++e) {
        const int K    = kt * 32 + (e & 7) + ((e >> 3) << 4) + 8 * lh;
        const int sect = K / D_;
        const int d    = K - sect * D_;
        const float zj = sZj[d];
        float v;
        if      (sect == 0) v = zj;                         // Zj
        else if (sect == 1) v = zi[d];                      // Zi
        else if (sect == 2) v = zj * zi[d];                 // Zj*Zi
        else                v = fabsf(zj - zi[d]);          // |Zj-Zi|
        a.s[e] = f2bf(v);
      }
#pragma unroll
      for (int nt = 0; nt < 6; ++nt) {
        BFrag bf;
        const int idx = ((kt * 6 + nt) * 32 + lane) * 2;    // uint4 units
        bf.q[0] = sW1[idx];
        bf.q[1] = sW1[idx + 1];
        acc[nt] = __builtin_amdgcn_wmma_f32_16x16x32_bf16(
            false, a.v, false, bf.v, (short)0, acc[nt], false, false);
      }
    }

    // logit[m] = sum_n relu(h[m,n]+b1[n]) * W2[n]
    // C/D layout: VGPR v holds M = v + 8*(lane/16), N = lane%16 (+16*nt)
    float part[8];
#pragma unroll
    for (int v = 0; v < 8; ++v) part[v] = 0.f;
#pragma unroll
    for (int nt = 0; nt < 6; ++nt) {
      const int n = nt * 16 + ln;
      const float b1v = b1[n], w2v = W2[n];
#pragma unroll
      for (int v = 0; v < 8; ++v)
        part[v] += fmaxf(acc[nt][v] + b1v, 0.f) * w2v;
    }
#pragma unroll
    for (int v = 0; v < 8; ++v) {        // reduce the 16 N-lanes per half
      part[v] += __shfl_xor(part[v], 1, 32);
      part[v] += __shfl_xor(part[v], 2, 32);
      part[v] += __shfl_xor(part[v], 4, 32);
      part[v] += __shfl_xor(part[v], 8, 32);
    }
    if (ln == 0) {
#pragma unroll
      for (int v = 0; v < 8; ++v) slog[i0 + 8 * lh + v] = part[v];
    }
  }
  __syncthreads();

  // ---- softmax over 512 keys (additive key mask + b2), LDS resident ----
  const float b2v = b2[0];
  const float* mrow = mask + (size_t)b * S_;
  float l0 = slog[tid]       + b2v + (1.f - mrow[tid])       * -3.4028235e38f;
  float l1 = slog[tid + 256] + b2v + (1.f - mrow[tid + 256]) * -3.4028235e38f;
  slog[tid] = l0; slog[tid + 256] = l1;
  sred[tid] = fmaxf(l0, l1);
  __syncthreads();
  for (int off = 128; off > 0; off >>= 1) {
    if (tid < off) sred[tid] = fmaxf(sred[tid], sred[tid + off]);
    __syncthreads();
  }
  const float maxv = sred[0];
  __syncthreads();
  const float e0 = __expf(l0 - maxv), e1 = __expf(l1 - maxv);
  slog[tid] = e0; slog[tid + 256] = e1;
  sred[tid] = e0 + e1;
  __syncthreads();
  for (int off = 128; off > 0; off >>= 1) {
    if (tid < off) sred[tid] += sred[tid + off];
    __syncthreads();
  }
  const float inv = 1.f / sred[0];

  // ---- ctx[j,:] = sum_i probs[i] * H_i[b,i,:]  (L2-resident streaming) ----
  float a0 = 0.f, a1 = 0.f, a2 = 0.f, a3 = 0.f;
  const float* Hib = Hi + (size_t)b * S_ * H_;
  for (int i = 0; i < S_; ++i) {
    const float p = slog[i];
    const float* r = Hib + (size_t)i * H_;
    a0 += p * r[tid];
    a1 += p * r[tid + 256];
    a2 += p * r[tid + 512];
    a3 += p * r[tid + 768];
  }
  const size_t ob = (size_t)rowj * H_;
  ctxb[ob + tid]       = f2bf(a0 * inv);
  ctxb[ob + tid + 256] = f2bf(a1 * inv);
  ctxb[ob + tid + 512] = f2bf(a2 * inv);
  ctxb[ob + tid + 768] = f2bf(a3 * inv);
}

// ---------------------------------------------------------------------------
// Kernel 4: msg1 = relu([ctx, H_j] @ Wv1 + bv1), bf16 out.
// M=1024, N=1024, K=2048. 128x128 block tile, 16x128 per wave.
// Weight slab (8KB per ktile, shared by all 8 waves) staged via TDM, double
// buffered: wave0 issues TENSOR_LOAD_TO_LDS for kt+1 while everyone computes kt.
// ---------------------------------------------------------------------------
__global__ void __launch_bounds__(256)
gemm1_kernel(const unsigned short* __restrict__ ctxb,
             const unsigned short* __restrict__ Hjb,
             const unsigned short* __restrict__ Wp,
             const float* __restrict__ bv1,
             unsigned short* __restrict__ msg1) {
#if USE_TDM
  __shared__ u32x4 sB[2][512];                  // 2 x 8KB weight slabs
#endif
  const int tid  = threadIdx.x;
  const int wave = tid >> 5, lane = tid & 31, lh = lane >> 4, ln = lane & 15;
  const int bm = blockIdx.x >> 3, bn = blockIdx.x & 7;
  const int m  = bm * 128 + wave * 16 + ln;     // A row for this lane

  v8f acc[8];
  const v8f vz = {0.f, 0.f, 0.f, 0.f, 0.f, 0.f, 0.f, 0.f};
#pragma unroll
  for (int nt = 0; nt < 8; ++nt) acc[nt] = vz;

#if USE_TDM
  const unsigned lds0 = (unsigned)(size_t)(void*)&sB[0][0];
  const unsigned lds1 = (unsigned)(size_t)(void*)&sB[1][0];
  if (wave == 0)
    tdm_load_1d(lds0, Wp + ((size_t)bn * 8) * 512, 1024);   // kt=0 slab
#endif

  for (int kt = 0; kt < 64; ++kt) {
    // A = concat(ctx, H_j) in bf16; 32-wide k-tiles never straddle the seam
    const unsigned short* Ab;
    int kk;
    if (kt < 32) { Ab = ctxb; kk = kt * 32; }
    else         { Ab = Hjb;  kk = kt * 32 - 1024; }
    BFrag a;
    const unsigned short* ap = Ab + (size_t)m * H_ + kk + 8 * lh;
    a.q[0] = *(const u32x4*)(ap);
    a.q[1] = *(const u32x4*)(ap + 16);

#if USE_TDM
    if (wave == 0) __builtin_amdgcn_s_wait_tensorcnt(0);    // slab kt landed
    __syncthreads();                                        // publish to all waves
    if (wave == 0 && kt + 1 < 64)                           // prefetch kt+1 slab
      tdm_load_1d((kt & 1) ? lds0 : lds1,
                  Wp + ((size_t)(kt + 1) * 64 + bn * 8) * 512, 1024);
    const u32x4* bbuf = sB[kt & 1];
#pragma unroll
    for (int nt = 0; nt < 8; ++nt) {
      BFrag bf;
      bf.q[0] = bbuf[(nt * 32 + lane) * 2];
      bf.q[1] = bbuf[(nt * 32 + lane) * 2 + 1];
      acc[nt] = __builtin_amdgcn_wmma_f32_16x16x32_bf16(
          false, a.v, false, bf.v, (short)0, acc[nt], false, false);
    }
#else
    if (kt + 1 < 64)
      __builtin_prefetch(Wp + (((size_t)(kt + 1) * 64 + bn * 8) * 32 + lane) * 16, 0, 0);
#pragma unroll
    for (int nt = 0; nt < 8; ++nt) {
      BFrag bf;
      const unsigned short* bp =
          Wp + (((size_t)kt * 64 + (bn * 8 + nt)) * 32 + lane) * 16;
      bf.q[0] = *(const u32x4*)bp;
      bf.q[1] = *(const u32x4*)(bp + 8);
      acc[nt] = __builtin_amdgcn_wmma_f32_16x16x32_bf16(
          false, a.v, false, bf.v, (short)0, acc[nt], false, false);
    }
#endif
  }

  const int mb = bm * 128 + wave * 16;
#pragma unroll
  for (int nt = 0; nt < 8; ++nt) {
    const int n = bn * 128 + nt * 16 + ln;
    const float bb = bv1[n];
#pragma unroll
    for (int v = 0; v < 8; ++v) {
      const int mr = mb + v + 8 * lh;
      msg1[(size_t)mr * OUT_HID + n] = f2bf(fmaxf(acc[nt][v] + bb, 0.f));
    }
  }
}

// ---------------------------------------------------------------------------
// Kernel 5: out = alpha * (msg1 @ Wv2 + bv2), fp32 out. M=N=K=1024.
// Same TDM double-buffered weight staging.
// ---------------------------------------------------------------------------
__global__ void __launch_bounds__(256)
gemm2_kernel(const unsigned short* __restrict__ msg1,
             const unsigned short* __restrict__ Wp,
             const float* __restrict__ bv2, const float* __restrict__ alpha,
             float* __restrict__ out) {
#if USE_TDM
  __shared__ u32x4 sB[2][512];
#endif
  const int tid  = threadIdx.x;
  const int wave = tid >> 5, lane = tid & 31, lh = lane >> 4, ln = lane & 15;
  const int bm = blockIdx.x >> 3, bn = blockIdx.x & 7;
  const int m  = bm * 128 + wave * 16 + ln;

  v8f acc[8];
  const v8f vz = {0.f, 0.f, 0.f, 0.f, 0.f, 0.f, 0.f, 0.f};
#pragma unroll
  for (int nt = 0; nt < 8; ++nt) acc[nt] = vz;

#if USE_TDM
  const unsigned lds0 = (unsigned)(size_t)(void*)&sB[0][0];
  const unsigned lds1 = (unsigned)(size_t)(void*)&sB[1][0];
  if (wave == 0)
    tdm_load_1d(lds0, Wp + ((size_t)bn * 8) * 512, 1024);
#endif

  for (int kt = 0; kt < 32; ++kt) {
    BFrag a;
    const unsigned short* ap = msg1 + (size_t)m * OUT_HID + kt * 32 + 8 * lh;
    a.q[0] = *(const u32x4*)(ap);
    a.q[1] = *(const u32x4*)(ap + 16);

#if USE_TDM
    if (wave == 0) __builtin_amdgcn_s_wait_tensorcnt(0);
    __syncthreads();
    if (wave == 0 && kt + 1 < 32)
      tdm_load_1d((kt & 1) ? lds0 : lds1,
                  Wp + ((size_t)(kt + 1) * 64 + bn * 8) * 512, 1024);
    const u32x4* bbuf = sB[kt & 1];
#pragma unroll
    for (int nt = 0; nt < 8; ++nt) {
      BFrag bf;
      bf.q[0] = bbuf[(nt * 32 + lane) * 2];
      bf.q[1] = bbuf[(nt * 32 + lane) * 2 + 1];
      acc[nt] = __builtin_amdgcn_wmma_f32_16x16x32_bf16(
          false, a.v, false, bf.v, (short)0, acc[nt], false, false);
    }
#else
#pragma unroll
    for (int nt = 0; nt < 8; ++nt) {
      BFrag bf;
      const unsigned short* bp =
          Wp + (((size_t)kt * 64 + (bn * 8 + nt)) * 32 + lane) * 16;
      bf.q[0] = *(const u32x4*)bp;
      bf.q[1] = *(const u32x4*)(bp + 8);
      acc[nt] = __builtin_amdgcn_wmma_f32_16x16x32_bf16(
          false, a.v, false, bf.v, (short)0, acc[nt], false, false);
    }
#endif
  }

  const float al = alpha[0];
  const int mb = bm * 128 + wave * 16;
#pragma unroll
  for (int nt = 0; nt < 8; ++nt) {
    const int n = bn * 128 + nt * 16 + ln;
    const float bb = bv2[n];
#pragma unroll
    for (int v = 0; v < 8; ++v) {
      const int mr = mb + v + 8 * lh;
      out[(size_t)mr * H_ + n] = al * (acc[nt][v] + bb);
    }
  }
}

// ---------------------------------------------------------------------------
extern "C" void kernel_launch(void* const* d_in, const int* in_sizes, int n_in,
                              void* d_out, int out_size, void* d_ws, size_t ws_size,
                              hipStream_t stream) {
  (void)in_sizes; (void)n_in; (void)out_size; (void)ws_size;

  const float* Hj    = (const float*)d_in[0];
  const float* Hi    = (const float*)d_in[1];
  const float* mask  = (const float*)d_in[2];
  const float* pj    = (const float*)d_in[3];
  const float* pi    = (const float*)d_in[4];
  const float* W1    = (const float*)d_in[5];
  const float* b1    = (const float*)d_in[6];
  const float* W2    = (const float*)d_in[7];
  const float* b2    = (const float*)d_in[8];
  const float* Wv1   = (const float*)d_in[9];
  const float* bv1   = (const float*)d_in[10];
  const float* Wv2   = (const float*)d_in[11];
  const float* bv2   = (const float*)d_in[12];
  const float* alpha = (const float*)d_in[13];

  // workspace carve-out (~12.3 MB total)
  char* ws = (char*)d_ws;
  size_t off = 0;
  auto take = [&](size_t bytes) -> void* {
    void* p = ws + off;
    off = (off + bytes + 255) & ~(size_t)255;
    return p;
  };
  float*          Zj    = (float*)take((size_t)B_ * S_ * D_ * 4);
  float*          Zi    = (float*)take((size_t)B_ * S_ * D_ * 4);
  unsigned short* W1p   = (unsigned short*)take((size_t)96 * 96 * 2);
  unsigned short* Wv1p  = (unsigned short*)take((size_t)VAL_IN * OUT_HID * 2);
  unsigned short* Wv2p  = (unsigned short*)take((size_t)OUT_HID * H_ * 2);
  unsigned short* Hjb   = (unsigned short*)take((size_t)B_ * S_ * H_ * 2);
  unsigned short* ctxb  = (unsigned short*)take((size_t)B_ * S_ * H_ * 2);
  unsigned short* msg1  = (unsigned short*)take((size_t)B_ * S_ * OUT_HID * 2);

  // 1) projections + bf16(H_j)
  proj_kernel<<<B_ * S_, 256, 0, stream>>>(Hj, Hi, pj, pi, Zj, Zi, Hjb);

  // 2) weight repacks into WMMA B-fragment order
  pack_kernel<<<3,   256, 0, stream>>>(W1,  W1p,  96,      96);
  pack_kernel<<<512, 256, 0, stream>>>(Wv1, Wv1p, VAL_IN,  OUT_HID);
  pack_kernel<<<256, 256, 0, stream>>>(Wv2, Wv2p, OUT_HID, H_);

  // 3) fused pair-MLP + softmax + context (logits never leave LDS)
  pair_kernel<<<B_ * S_, 256, 0, stream>>>(Zj, Zi, W1p, b1, W2, b2, mask, Hi, ctxb);

  // 4) value-fusion MLP, two WMMA GEMMs with TDM-staged weights
  gemm1_kernel<<<64, 256, 0, stream>>>(ctxb, Hjb, Wv1p, bv1, msg1);
  gemm2_kernel<<<64, 256, 0, stream>>>(msg1, Wv2p, bv2, alpha, (float*)d_out);
}
